// rgChromaHistBlock_75453985456768
// MI455X (gfx1250) — matrix operand order repaired and checked by reading, not compiled, not run on hardware
//
#include <hip/hip_runtime.h>
#include <stdint.h>

typedef _Float16 v16h __attribute__((ext_vector_type(16)));
typedef _Float16 v8h  __attribute__((ext_vector_type(8)));
typedef float    v8f  __attribute__((ext_vector_type(8)));
typedef v8h      v8h_a  __attribute__((may_alias));
typedef uint32_t u32_a  __attribute__((may_alias));
typedef float    f32_a  __attribute__((may_alias));

#define H_BINS        64
#define NPIX          (448 * 448)   // 200704 pixels per batch
#define NBATCH        16
#define WGS_PER_BATCH 32
#define STEPS         (NPIX / (WGS_PER_BATCH * 32))  // 196 K-steps of 32 pixels (even)
#define PITCH         40            // f16 per bin row: 32 pixels + 8 pad (80B, 16B aligned)
#define BUF           (H_BINS * PITCH)               // 2560 f16 = 5120 B per buffer

__global__ __launch_bounds__(256) void zeroAcc_75453985456768(float* acc, int n) {
    int i = blockIdx.x * 256 + threadIdx.x;
    if (i < n) acc[i] = 0.0f;
}

__global__ __launch_bounds__(256) void rgHistWmma_75453985456768(
        const float* __restrict__ x, float* __restrict__ acc) {
    // double-buffered staging: [buf][bin][pixel] f16
    __shared__ __align__(16) _Float16 s_kr[2 * BUF];
    __shared__ __align__(16) _Float16 s_kg[2 * BUF];

    const int b    = blockIdx.y;
    const int wg   = blockIdx.x;         // 0..31 within batch
    const int tid  = threadIdx.x;
    const int lane = tid & 31;           // wave32
    const int wave = tid >> 5;           // 0..7

    const float* xb = x + (size_t)b * 3 * NPIX;

    // ---- producer mapping: pixel pair x 4 bins per thread (2048 vals/channel) ----
    const int pLane   = (lane & 15) * 2;                 // pixel pair base 0,2,..,30
    const int pBin    = wave * 8 + (lane >> 4) * 4;      // 4 bins starting here
    const float inv63 = 1.0f / 63.0f;

    // ---- consumer mapping: 4x4 tiles of 16x16, 2 tiles per wave ----
    const int tr  = wave >> 1;           // r-tile 0..3
    const int tg0 = (wave & 1) * 2;      // g-tile base: 0 or 2
    const int hs  = lane >> 4;           // half select (lanes 0-15 vs 16-31)
    const int l15 = lane & 15;

    v8f acc0 = {};
    v8f acc1 = {};

    const int pixBase = wg * (STEPS * 32);

    // Producer: compute inverse-quadratic kernels for 2 pixels x 4 bins, packed b32 stores.
    auto produce = [&](int step, _Float16* kr, _Float16* kg) {
        const int pix0 = pixBase + step * 32 + pLane;
        const float2 rr = *(const float2*)(const f32_a*)(xb + pix0);
        const float2 gg = *(const float2*)(const f32_a*)(xb + NPIX + pix0);
        const float2 bb = *(const float2*)(const f32_a*)(xb + 2 * NPIX + pix0);
        const float r0 = fminf(fmaxf(rr.x, 0.f), 1.f), r1 = fminf(fmaxf(rr.y, 0.f), 1.f);
        const float g0 = fminf(fmaxf(gg.x, 0.f), 1.f), g1 = fminf(fmaxf(gg.y, 0.f), 1.f);
        const float b0 = fminf(fmaxf(bb.x, 0.f), 1.f), b1 = fminf(fmaxf(bb.y, 0.f), 1.f);
        const float i0 = __builtin_amdgcn_rcpf(r0 + g0 + b0 + 1e-6f);
        const float i1 = __builtin_amdgcn_rcpf(r1 + g1 + b1 + 1e-6f);
        const float Ir0 = r0 * i0, Ig0 = g0 * i0;
        const float Ir1 = r1 * i1, Ig1 = g1 * i1;
        #pragma unroll
        for (int j2 = 0; j2 < 4; ++j2) {
            const int   j = pBin + j2;
            const float c = (float)j * inv63;
            const float dr0 = (Ir0 - c) * 50.0f, dr1 = (Ir1 - c) * 50.0f;  // 1/SIGMA
            const float dg0 = (Ig0 - c) * 50.0f, dg1 = (Ig1 - c) * 50.0f;
            union { _Float16 h[2]; uint32_t u; } pr, pg;
            pr.h[0] = (_Float16)__builtin_amdgcn_rcpf(__builtin_fmaf(dr0, dr0, 1.0f));
            pr.h[1] = (_Float16)__builtin_amdgcn_rcpf(__builtin_fmaf(dr1, dr1, 1.0f));
            pg.h[0] = (_Float16)__builtin_amdgcn_rcpf(__builtin_fmaf(dg0, dg0, 1.0f));
            pg.h[1] = (_Float16)__builtin_amdgcn_rcpf(__builtin_fmaf(dg1, dg1, 1.0f));
            *(u32_a*)&kr[j * PITCH + pLane] = pr.u;   // 4B-aligned, 32 distinct banks
            *(u32_a*)&kg[j * PITCH + pLane] = pg.u;
        }
    };

    // Consumer: one A fragment, two B fragments, two WMMAs (rank-32 update).
    auto consume = [&](const _Float16* kr, const _Float16* kg) {
        // A (Kr^T, 16x32 f16): lanes 0-15 K 0-15; lanes 16-31 K 16-31 (contiguous).
        const _Float16* arow = &kr[(tr * 16 + l15) * PITCH];
        union { v16h v; v8h h[2]; } A, B0, B1;
        A.h[0] = *(const v8h_a*)(arow + hs * 16);
        A.h[1] = *(const v8h_a*)(arow + hs * 16 + 8);
        // B (Kg, 32x16 f16): lanes 0-15 K 0-15; lanes 16-31 K 16-31 (contiguous).
        const _Float16* bcol0 = &kg[(tg0 * 16 + l15) * PITCH] + hs * 16;
        const _Float16* bcol1 = &kg[((tg0 + 1) * 16 + l15) * PITCH] + hs * 16;
        B0.h[0] = *(const v8h_a*)(bcol0);
        B0.h[1] = *(const v8h_a*)(bcol0 + 8);
        B1.h[0] = *(const v8h_a*)(bcol1);
        B1.h[1] = *(const v8h_a*)(bcol1 + 8);
        acc0 = __builtin_amdgcn_wmma_f32_16x16x32_f16(
                   false, A.v, false, B0.v, (short)0, acc0, false, false);
        acc1 = __builtin_amdgcn_wmma_f32_16x16x32_f16(
                   false, A.v, false, B1.v, (short)0, acc1, false, false);
    };

    // Unrolled by 2: compile-time buffer offsets, one barrier per K-step.
    for (int step = 0; step < STEPS; step += 2) {
        produce(step, s_kr, s_kg);
        __syncthreads();
        consume(s_kr, s_kg);

        produce(step + 1, s_kr + BUF, s_kg + BUF);
        __syncthreads();
        consume(s_kr + BUF, s_kg + BUF);
    }

    // ---- fold partial tiles into per-batch accumulator --------------------------
    // C/D layout: lane n<16 -> (M=i, N=n); lane n+16 -> (M=8+i, N=n).
    float* hacc = acc + (size_t)b * (H_BINS * H_BINS);
    #pragma unroll
    for (int i = 0; i < 8; ++i) {
        const int row = tr * 16 + hs * 8 + i;
        const int col = tg0 * 16 + l15;
        atomicAdd(&hacc[row * H_BINS + col],      acc0[i]);
        atomicAdd(&hacc[row * H_BINS + col + 16], acc1[i]);
    }
}

__global__ __launch_bounds__(256) void normHist_75453985456768(
        const float* __restrict__ acc, float* __restrict__ out) {
    __shared__ float red[256];
    const int b   = blockIdx.x;
    const int tid = threadIdx.x;
    const float* hacc = acc + (size_t)b * (H_BINS * H_BINS);
    float s = 0.0f;
    for (int i = tid; i < H_BINS * H_BINS; i += 256) s += hacc[i];
    red[tid] = s;
    __syncthreads();
    for (int off = 128; off > 0; off >>= 1) {
        if (tid < off) red[tid] += red[tid + off];
        __syncthreads();
    }
    const float invn = 1.0f / (red[0] + 1e-6f);
    float* ob = out + (size_t)b * (H_BINS * H_BINS);
    for (int i = tid; i < H_BINS * H_BINS; i += 256) ob[i] = hacc[i] * invn;
}

extern "C" void kernel_launch(void* const* d_in, const int* in_sizes, int n_in,
                              void* d_out, int out_size, void* d_ws, size_t ws_size,
                              hipStream_t stream) {
    (void)in_sizes; (void)n_in; (void)out_size; (void)ws_size;
    const float* x   = (const float*)d_in[0];
    float*       out = (float*)d_out;
    float*       acc = (float*)d_ws;       // 16 * 64 * 64 floats = 256 KB scratch

    const int accN = NBATCH * H_BINS * H_BINS;
    zeroAcc_75453985456768<<<(accN + 255) / 256, 256, 0, stream>>>(acc, accN);

    dim3 grid(WGS_PER_BATCH, NBATCH);
    rgHistWmma_75453985456768<<<grid, 256, 0, stream>>>(x, acc);

    normHist_75453985456768<<<NBATCH, 256, 0, stream>>>(acc, out);
}
